// SearchTransfer2_tmp_49452253446912
// MI455X (gfx1250) — compile-verified, matrix operand order checked by reading, not run on hardware
//
#include <hip/hip_runtime.h>
#include <math.h>

// ---------------------------------------------------------------------------
// CDNA5 (gfx1250) WMMA types
// ---------------------------------------------------------------------------
typedef __attribute__((ext_vector_type(16))) _Float16 v16h;
typedef __attribute__((ext_vector_type(8)))  _Float16 v8h;
typedef __attribute__((ext_vector_type(8)))  float    v8f;

__device__ __forceinline__ float gelu_exact(float x) {
    // exact gelu: 0.5*x*(1+erf(x/sqrt(2)))
    return 0.5f * x * (1.0f + erff(x * 0.70710678118654752f));
}

// ---------------------------------------------------------------------------
// conv3x3 (SAME) + bias + exact GELU, scattering result as f16 directly into
// the unfolded layout (fusing F.unfold, no f32 intermediate tensor):
//   mode 0: dst[((b*1024 + l)*4096) + cc]   "A" layout  [b][L][CK]
//   mode 1: dst[((b*4096 + cc)*1024) + l]   "Bt" layout [b][CK][L]  (V^T)
// where l = (h/8)*32 + (w/8), cc = c*64 + (h%8)*8 + (w%8)
// ---------------------------------------------------------------------------
__global__ void conv3x3_gelu_scatter(const float* __restrict__ x,
                                     const float* __restrict__ wgt,
                                     const float* __restrict__ bias,
                                     _Float16* __restrict__ dst, int mode)
{
    int idx = blockIdx.x * blockDim.x + threadIdx.x;     // B*C*H*W = 16777216
    int w = idx & 255;
    int h = (idx >> 8) & 255;
    int c = (idx >> 16) & 63;
    int b = idx >> 22;
    float acc = bias[c];
    const float* xb = x + ((size_t)b * 64) * 65536;
    const float* wc = wgt + (size_t)c * 576;
    #pragma unroll 1
    for (int ci = 0; ci < 64; ++ci) {
        const float* xp = xb + (size_t)ci * 65536;
        const float* wp = wc + ci * 9;
        #pragma unroll
        for (int dy = -1; dy <= 1; ++dy) {
            int hh = h + dy;
            if (hh < 0 || hh > 255) continue;
            #pragma unroll
            for (int dx = -1; dx <= 1; ++dx) {
                int ww = w + dx;
                if (ww < 0 || ww > 255) continue;
                acc += xp[hh * 256 + ww] * wp[(dy + 1) * 3 + (dx + 1)];
            }
        }
    }
    float y = gelu_exact(acc);
    int hb = h >> 3, kh = h & 7, wb = w >> 3, kw = w & 7;
    int l  = hb * 32 + wb;
    int cc = c * 64 + kh * 8 + kw;
    size_t off = (mode == 0) ? (((size_t)b * 1024 + l) * 4096 + cc)
                             : (((size_t)b * 4096 + cc) * 1024 + l);
    dst[off] = (_Float16)y;
}

// f32 -> f16 cast (FC weights; already [N][K] row-major, i.e. Bt layout)
__global__ void pack_f32_to_f16(const float* __restrict__ src,
                                _Float16* __restrict__ dst, int n)
{
    int i = blockIdx.x * blockDim.x + threadIdx.x;
    if (i < n) dst[i] = (_Float16)src[i];
}

// ---------------------------------------------------------------------------
// Register-blocked WMMA GEMM:  C[b] = scale * (A[b] x Bt[b]^T) + bias
//   A  : [M][K] f16 row-major (per-batch stride aBatch elements)
//   Bt : [N][K] f16 row-major (per-batch stride bBatch; 0 = shared weights)
//   C  : [M][N] f32 row-major
// One wave32 per 32x64 output tile: 2 A fragments x 4 B fragments feed
// 8 v_wmma_f32_16x16x32_f16 accumulators per K-step of 32 (12 b128 loads
// per 8 WMMAs).  Fragment layout per ISA 7.12.2: for lane = g*16+m the
// 16 halfs split into two contiguous 8-half (16B) chunks at K offsets
// k0+g*8 and k0+16+g*8; B mirrors with columns; C vgpr r holds element
// (r+8g, lane&15).  M%32==0, N%64==0, K%32==0 for all call sites.
// ---------------------------------------------------------------------------
union Frag { v16h v; v8h h[2]; };

__global__ __launch_bounds__(256)
void wmma_gemm_ABt(const _Float16* __restrict__ A, size_t aBatch,
                   const _Float16* __restrict__ Bt, size_t bBatch,
                   float* __restrict__ C, size_t cBatch,
                   int M, int N, int K,
                   const float* __restrict__ bias, float scale)
{
    int lane = threadIdx.x & 31;
    int wave = threadIdx.x >> 5;
    int tilesN = N >> 6;                         // 64-wide tiles
    int tile = blockIdx.x * (blockDim.x >> 5) + wave;
    int totTiles = (M >> 5) * tilesN;
    if (tile >= totTiles) return;                // wave-uniform
    int ti = tile / tilesN;                      // 32-row tile
    int tj = tile % tilesN;                      // 64-col tile
    int b = blockIdx.z;

    const _Float16* Ab  = A  + (size_t)b * aBatch;
    const _Float16* Btb = Bt + (size_t)b * bBatch;
    float*          Cb  = C  + (size_t)b * cBatch;

    int g  = lane >> 4;                          // K-half group
    int lm = lane & 15;                          // row-in-fragment
    int mBase = ti << 5;
    int nBase = tj << 6;

    const _Float16* arow[2];
    arow[0] = Ab + ((size_t)mBase + lm) * K;
    arow[1] = arow[0] + (size_t)16 * K;
    const _Float16* brow[4];
    #pragma unroll
    for (int j = 0; j < 4; ++j)
        brow[j] = Btb + ((size_t)(nBase + j * 16) + lm) * K;

    v8f acc[2][4] = {};
    for (int k0 = 0; k0 < K; k0 += 32) {
        // prefetch next K-block of the A streams into cache (global_prefetch_b8)
        if (k0 + 64 < K) {
            __builtin_prefetch(arow[0] + k0 + 64, 0, 0);
            __builtin_prefetch(arow[1] + k0 + 64, 0, 0);
        }
        int lo = k0 + g * 8;
        int hi = k0 + 16 + g * 8;
        Frag af[2], bf[4];
        #pragma unroll
        for (int i = 0; i < 2; ++i) {
            af[i].h[0] = *(const v8h*)(arow[i] + lo);
            af[i].h[1] = *(const v8h*)(arow[i] + hi);
        }
        #pragma unroll
        for (int j = 0; j < 4; ++j) {
            bf[j].h[0] = *(const v8h*)(brow[j] + lo);
            bf[j].h[1] = *(const v8h*)(brow[j] + hi);
        }
        #pragma unroll
        for (int i = 0; i < 2; ++i)
            #pragma unroll
            for (int j = 0; j < 4; ++j)
                acc[i][j] = __builtin_amdgcn_wmma_f32_16x16x32_f16(
                    /*neg_a=*/false, af[i].v, /*neg_b=*/false, bf[j].v,
                    /*c_mod=*/(short)0, acc[i][j],
                    /*reuse_a=*/false, /*reuse_b=*/false);
    }

    #pragma unroll
    for (int j = 0; j < 4; ++j) {
        int n = nBase + j * 16 + lm;
        float bb = bias ? bias[n] : 0.0f;
        #pragma unroll
        for (int i = 0; i < 2; ++i) {
            #pragma unroll
            for (int r = 0; r < 8; ++r) {
                int m = mBase + i * 16 + r + 8 * g;
                Cb[(size_t)m * N + n] = acc[i][j][r] * scale + bb;
            }
        }
    }
}

// ---------------------------------------------------------------------------
// Train-mode BatchNorm over channel d, stats across (B,L)=4096 samples,
// emitting normalized f16 (ready to be a WMMA operand).
// x,y layout: [BL][D] row-major. One block per channel d.
// ---------------------------------------------------------------------------
__global__ void bn_to_f16(const float* __restrict__ x,
                          const float* __restrict__ gamma,
                          const float* __restrict__ beta,
                          _Float16* __restrict__ y, int BL, int D)
{
    int d = blockIdx.x;
    __shared__ float red[256];
    float s = 0.f, s2 = 0.f;
    for (int i = threadIdx.x; i < BL; i += blockDim.x) {
        float v = x[(size_t)i * D + d];
        s += v; s2 += v * v;
    }
    red[threadIdx.x] = s; __syncthreads();
    for (int o = 128; o > 0; o >>= 1) {
        if (threadIdx.x < o) red[threadIdx.x] += red[threadIdx.x + o];
        __syncthreads();
    }
    float mean = red[0] / BL; __syncthreads();
    red[threadIdx.x] = s2; __syncthreads();
    for (int o = 128; o > 0; o >>= 1) {
        if (threadIdx.x < o) red[threadIdx.x] += red[threadIdx.x + o];
        __syncthreads();
    }
    float var = red[0] / BL - mean * mean;
    float gm = gamma[d] * rsqrtf(var + 1e-5f), bt = beta[d];
    for (int i = threadIdx.x; i < BL; i += blockDim.x) {
        float v = (x[(size_t)i * D + d] - mean) * gm + bt;
        y[(size_t)i * D + d] = (_Float16)v;
    }
}

// ---------------------------------------------------------------------------
// Row softmax on f32 scores, emitting f16 probabilities (next WMMA operand).
// ---------------------------------------------------------------------------
__global__ void softmax_rows_f16(const float* __restrict__ S,
                                 _Float16* __restrict__ P, int N)
{
    size_t row = blockIdx.x;
    const float* r = S + row * N;
    _Float16* o = P + row * N;
    __shared__ float red[256];
    float mx = -1e30f;
    for (int i = threadIdx.x; i < N; i += blockDim.x) mx = fmaxf(mx, r[i]);
    red[threadIdx.x] = mx; __syncthreads();
    for (int s = 128; s > 0; s >>= 1) {
        if (threadIdx.x < s) red[threadIdx.x] = fmaxf(red[threadIdx.x], red[threadIdx.x + s]);
        __syncthreads();
    }
    mx = red[0]; __syncthreads();
    float sum = 0.f;
    for (int i = threadIdx.x; i < N; i += blockDim.x) sum += __expf(r[i] - mx);
    red[threadIdx.x] = sum; __syncthreads();
    for (int s = 128; s > 0; s >>= 1) {
        if (threadIdx.x < s) red[threadIdx.x] += red[threadIdx.x + s];
        __syncthreads();
    }
    float inv = 1.0f / red[0];
    for (int i = threadIdx.x; i < N; i += blockDim.x)
        o[i] = (_Float16)(__expf(r[i] - mx) * inv);
}

// ---------------------------------------------------------------------------
// Fused fold + FF1(gelu) + FF2(gelu) + residual.  One block per pixel:
// thread o computes hidden[o] (o<256), then threads t<64 compute outputs.
// cols: [B][1024][4096] f32 (attn @ V, un-transposed unfold layout).
// ---------------------------------------------------------------------------
__global__ void ff_fold_kernel(const float* __restrict__ cols,
                               const float* __restrict__ ff1w, const float* __restrict__ ff1b,
                               const float* __restrict__ ff2w, const float* __restrict__ ff2b,
                               const float* __restrict__ feat, float* __restrict__ out)
{
    int p = blockIdx.x;                       // B*H*W = 262144
    int w = p & 255, h = (p >> 8) & 255, b = p >> 16;
    int hb = h >> 3, kh = h & 7, wb = w >> 3, kw = w & 7;
    int l = hb * 32 + wb;
    __shared__ float xs[64];
    __shared__ float hs[256];
    int t = threadIdx.x;
    if (t < 64)
        xs[t] = cols[((size_t)b * 1024 + l) * 4096 + t * 64 + kh * 8 + kw];
    __syncthreads();
    float s = ff1b[t];
    #pragma unroll 8
    for (int c = 0; c < 64; ++c) s += xs[c] * ff1w[t * 64 + c];
    hs[t] = gelu_exact(s);
    __syncthreads();
    if (t < 64) {
        float s2 = ff2b[t];
        #pragma unroll 8
        for (int j = 0; j < 256; ++j) s2 += hs[j] * ff2w[t * 256 + j];
        size_t off = ((size_t)b * 64 + t) * 65536 + (size_t)h * 256 + w;
        out[off] = gelu_exact(s2) + feat[off];
    }
}

// ---------------------------------------------------------------------------
extern "C" void kernel_launch(void* const* d_in, const int* in_sizes, int n_in,
                              void* d_out, int out_size, void* d_ws, size_t ws_size,
                              hipStream_t stream)
{
    const float* feat  = (const float*)d_in[0];
    const float* w_enc = (const float*)d_in[1];  const float* b_enc = (const float*)d_in[2];
    const float* w_k   = (const float*)d_in[3];  const float* b_k   = (const float*)d_in[4];
    const float* w_v   = (const float*)d_in[5];  const float* b_v   = (const float*)d_in[6];
    const float* fcq_w = (const float*)d_in[7];  const float* fcq_b = (const float*)d_in[8];
    const float* fck_w = (const float*)d_in[9];  const float* fck_b = (const float*)d_in[10];
    const float* bnq_g = (const float*)d_in[11]; const float* bnq_b = (const float*)d_in[12];
    const float* bnk_g = (const float*)d_in[13]; const float* bnk_b = (const float*)d_in[14];
    const float* ff1_w = (const float*)d_in[15]; const float* ff1_b = (const float*)d_in[16];
    const float* ff2_w = (const float*)d_in[17]; const float* ff2_b = (const float*)d_in[18];
    float* out = (float*)d_out;

    // workspace partition (bytes, 256-aligned)
    char* ws = (char*)d_ws;
    size_t off = 0;
    auto take = [&](size_t bytes) { char* p = ws + off; off = (off + bytes + 255) & ~(size_t)255; return p; };
    _Float16* Uq   = (_Float16*)take((size_t)4 * 1024 * 4096 * 2);  // unfolded refer (A)
    _Float16* Uk   = (_Float16*)take((size_t)4 * 1024 * 4096 * 2);  // unfolded fk    (A)
    _Float16* vT   = (_Float16*)take((size_t)4 * 4096 * 1024 * 2);  // unfolded fv^T  (Bt)
    _Float16* Wq16 = (_Float16*)take((size_t)1024 * 4096 * 2);      // fcq_w f16 (Bt)
    _Float16* Wk16 = (_Float16*)take((size_t)1024 * 4096 * 2);      // fck_w f16 (Bt)
    float*    qpre = (float*)take((size_t)4 * 1024 * 1024 * 4);
    float*    kpre = (float*)take((size_t)4 * 1024 * 1024 * 4);
    _Float16* q16  = (_Float16*)take((size_t)4 * 1024 * 1024 * 2);
    _Float16* k16  = (_Float16*)take((size_t)4 * 1024 * 1024 * 2);
    float*    scor = (float*)take((size_t)4 * 1024 * 1024 * 4);
    _Float16* attn = (_Float16*)take((size_t)4 * 1024 * 1024 * 2);
    float*    cols = (float*)take((size_t)4 * 1024 * 4096 * 4);

    // 1-3) convs + gelu, fused unfold scatter to f16
    conv3x3_gelu_scatter<<<65536, 256, 0, stream>>>(feat, w_enc, b_enc, Uq, 0);
    conv3x3_gelu_scatter<<<65536, 256, 0, stream>>>(feat, w_k,   b_k,   Uk, 0);
    conv3x3_gelu_scatter<<<65536, 256, 0, stream>>>(feat, w_v,   b_v,   vT, 1);

    // 4) cast FC weights to f16 (already [N][K] = Bt layout)
    pack_f32_to_f16<<<(4194304 + 255) / 256, 256, 0, stream>>>(fcq_w, Wq16, 4194304);
    pack_f32_to_f16<<<(4194304 + 255) / 256, 256, 0, stream>>>(fck_w, Wk16, 4194304);

    // 5) FC GEMMs: [4,1024,4096] x [1024,4096]^T -> [4,1024,1024]  (WMMA)
    //    wave-tiles = (1024/32)*(1024/64) = 512; 8 waves/block -> 64 blocks
    wmma_gemm_ABt<<<dim3(64, 1, 4), 256, 0, stream>>>(
        Uq, (size_t)1024 * 4096, Wq16, 0, qpre, (size_t)1024 * 1024,
        1024, 1024, 4096, fcq_b, 1.0f);
    wmma_gemm_ABt<<<dim3(64, 1, 4), 256, 0, stream>>>(
        Uk, (size_t)1024 * 4096, Wk16, 0, kpre, (size_t)1024 * 1024,
        1024, 1024, 4096, fck_b, 1.0f);

    // 6) BatchNorm (stats over B*L=4096) -> f16 operands
    bn_to_f16<<<1024, 256, 0, stream>>>(qpre, bnq_g, bnq_b, q16, 4096, 1024);
    bn_to_f16<<<1024, 256, 0, stream>>>(kpre, bnk_g, bnk_b, k16, 4096, 1024);

    // 7) scores = q k^T / 32   (WMMA, Bt = k naturally)
    wmma_gemm_ABt<<<dim3(64, 1, 4), 256, 0, stream>>>(
        q16, (size_t)1024 * 1024, k16, (size_t)1024 * 1024, scor, (size_t)1024 * 1024,
        1024, 1024, 1024, nullptr, 0.03125f);

    // 8) softmax rows -> f16 probs
    softmax_rows_f16<<<4096, 256, 0, stream>>>(scor, attn, 1024);

    // 9) out_cols = attn @ V   (WMMA, Bt = V^T from step 3)
    //    wave-tiles = (1024/32)*(4096/64) = 2048 -> 256 blocks
    wmma_gemm_ABt<<<dim3(256, 1, 4), 256, 0, stream>>>(
        attn, (size_t)1024 * 1024, vT, (size_t)4096 * 1024, cols, (size_t)1024 * 4096,
        1024, 4096, 1024, nullptr, 1.0f);

    // 10) fold + FF1 + FF2 + residual
    ff_fold_kernel<<<262144, 256, 0, stream>>>(cols, ff1_w, ff1_b, ff2_w, ff2_b, feat, out);

    (void)in_sizes; (void)n_in; (void)out_size; (void)ws_size;
}
